// HierarchicalClustering_12661563588794
// MI455X (gfx1250) — matrix operand order repaired and checked by reading, not compiled, not run on hardware
//
#include <hip/hip_runtime.h>
#include <hip/hip_bf16.h>
#include <math.h>

// ---------------------------------------------------------------------------
// CDNA5 (gfx1250) WMMA-based hierarchical clustering forward pass.
// Big GEMMs: v_wmma_f32_16x16x32_bf16, bf16-hi/lo (bf16x3) split precision,
// fp32 accumulators, (16*MB)x(16*NB) register blocking per wave.
// hi/lo interleaved per row ([hi(K) | lo(K)], stride 2K) so one pointer per
// tile serves all 4 loads; K is a template constant so lo-offsets fold into
// the 24-bit instruction offset (no extra address VGPRs).
// ---------------------------------------------------------------------------

typedef __bf16 bf16_t;
typedef __attribute__((ext_vector_type(16))) __bf16 v16bf;
typedef __attribute__((ext_vector_type(8)))  __bf16 v8bf;
typedef __attribute__((ext_vector_type(8)))  float  v8f;

union ABf { v16bf v; v8bf h[2]; };

#define WMMA_BF16(a, b, c) \
  __builtin_amdgcn_wmma_f32_16x16x32_bf16(false, (a), false, (b), (short)0, (c), false, false)

// NT block product: D(16*MB x 16*NB) += A_j(16xK) * B_i(16xK)^T.
// A/B rows are [hi(K)|lo(K)], row stride 2K.  CDNA5 VGPR layouts (CDNA5 ISA):
//   A 16-bit 16x32: lane L -> row (L&15); K chunks at g*8 and 16+g*8 (g=L>>4)
//   B 16-bit 32x16: lane L -> col (L&15); K chunks at g*8 and 16+g*8 (g=L>>4)
template <int MB, int NB, int K>
__device__ inline void block_mm_nt(const bf16_t* __restrict__ A,
                                   const bf16_t* __restrict__ B,
                                   int lane, v8f acc[MB][NB])
{
  const int r = lane & 15;
  const int g = lane >> 4;
  const bf16_t* pa[MB];
  #pragma unroll
  for (int j = 0; j < MB; ++j)
    pa[j] = A + ((size_t)(j * 16) + r) * (2 * K) + g * 8;
  const bf16_t* pb[NB];
  #pragma unroll
  for (int i = 0; i < NB; ++i)
    pb[i] = B + ((size_t)(i * 16) + r) * (2 * K) + g * 8;

  for (int k = 0; k < K; k += 32) {
    ABf ah[MB], al[MB];
    #pragma unroll
    for (int j = 0; j < MB; ++j) {
      ah[j].h[0] = *(const v8bf*)(pa[j] + k);
      ah[j].h[1] = *(const v8bf*)(pa[j] + k + 16);
      al[j].h[0] = *(const v8bf*)(pa[j] + K + k);      // lo: +K folds into imm
      al[j].h[1] = *(const v8bf*)(pa[j] + K + k + 16);
      __builtin_prefetch(pa[j] + k + 512, 0, 3);
    }
    #pragma unroll
    for (int i = 0; i < NB; ++i) {
      ABf bh, bl;
      bh.h[0] = *(const v8bf*)(pb[i] + k);
      bh.h[1] = *(const v8bf*)(pb[i] + k + 16);
      bl.h[0] = *(const v8bf*)(pb[i] + K + k);
      bl.h[1] = *(const v8bf*)(pb[i] + K + k + 16);
      #pragma unroll
      for (int j = 0; j < MB; ++j) {
        // bf16x3: hi*hi + hi*lo + lo*hi (lo*lo term ~2^-18, dropped)
        acc[j][i] = WMMA_BF16(ah[j].v, bh.v, acc[j][i]);
        acc[j][i] = WMMA_BF16(ah[j].v, bl.v, acc[j][i]);
        acc[j][i] = WMMA_BF16(al[j].v, bh.v, acc[j][i]);
      }
    }
  }
}

template <int MB, int NB>
__device__ inline void zero_acc(v8f acc[MB][NB])
{
  #pragma unroll
  for (int j = 0; j < MB; ++j)
    #pragma unroll
    for (int i = 0; i < NB; ++i)
      acc[j][i] = (v8f){0.f, 0.f, 0.f, 0.f, 0.f, 0.f, 0.f, 0.f};
}

// --- GEMM 1: logits S[i,j] = (2 h_i.c_j - |h_i|^2 - |c_j|^2) / (2*sigma) ----
template <int MB, int NB, int K>
__global__ void __launch_bounds__(256, 1)
gemm_logits_kernel(const bf16_t* __restrict__ Hhl, const bf16_t* __restrict__ Chl,
                   const float* __restrict__ hn, const float* __restrict__ cn,
                   float* __restrict__ S, int M, int N)
{
  const int wave = (int)((blockIdx.x * blockDim.x + threadIdx.x) >> 5);
  const int lane = (int)(threadIdx.x & 31);
  const int stripsN = (N >> 4) / NB;
  const int nBlocks = ((M >> 4) / MB) * stripsN;
  if (wave >= nBlocks) return;                 // wave-uniform: EXEC stays all-ones
  const int tm = wave / stripsN, tn = wave - tm * stripsN;
  const int row0 = tm * MB * 16, col0 = tn * NB * 16;

  v8f acc[MB][NB];
  zero_acc<MB, NB>(acc);
  block_mm_nt<MB, NB, K>(Hhl + (size_t)row0 * (2 * K),
                         Chl + (size_t)col0 * (2 * K), lane, acc);

  const int g = lane >> 4, n = lane & 15;
  #pragma unroll
  for (int j = 0; j < MB; ++j) {
    const int row = row0 + j * 16;
    #pragma unroll
    for (int i = 0; i < NB; ++i) {
      const int col = col0 + i * 16;
      #pragma unroll
      for (int v = 0; v < 8; ++v) {
        const int m = v + (g << 3);
        const float s = (2.0f * acc[j][i][v] - hn[row + m] - cn[col + n]) * 0.25f; // 1/(2*SIGMA)
        S[(size_t)(row + m) * N + (col + n)] = s;
      }
    }
  }
}

// --- GEMM 2: t = tanh(h W^T + b), stored TRANSPOSED, interleaved hi|lo -----
template <int MB, int NB, int K>
__global__ void __launch_bounds__(256, 1)
gemm_tanh_kernel(const bf16_t* __restrict__ Hhl, const bf16_t* __restrict__ Whl,
                 const float* __restrict__ bias,
                 bf16_t* __restrict__ tThl, int M, int N)
{
  const int wave = (int)((blockIdx.x * blockDim.x + threadIdx.x) >> 5);
  const int lane = (int)(threadIdx.x & 31);
  const int stripsN = (N >> 4) / NB;
  const int nBlocks = ((M >> 4) / MB) * stripsN;
  if (wave >= nBlocks) return;
  const int tm = wave / stripsN, tn = wave - tm * stripsN;
  const int row0 = tm * MB * 16, col0 = tn * NB * 16;

  v8f acc[MB][NB];
  zero_acc<MB, NB>(acc);
  block_mm_nt<MB, NB, K>(Hhl + (size_t)row0 * (2 * K),
                         Whl + (size_t)col0 * (2 * K), lane, acc);

  const int g = lane >> 4, n = lane & 15;
  #pragma unroll
  for (int j = 0; j < MB; ++j) {
    const int row = row0 + j * 16;
    #pragma unroll
    for (int i = 0; i < NB; ++i) {
      const int col = col0 + i * 16;
      #pragma unroll
      for (int v = 0; v < 8; ++v) {
        const int m = v + (g << 3);
        const float t = tanhf(acc[j][i][v] + bias[col + n]);
        const bf16_t hi = (bf16_t)t;
        const bf16_t lo = (bf16_t)(t - (float)hi);
        const size_t idx = (size_t)(col + n) * (2 * M) + (row + m); // [d, 2M] rows hi|lo
        tThl[idx] = hi;
        tThl[idx + M] = lo;
      }
    }
  }
}

// --- GEMM 3: h_next = p^T t  (A = pT [Ncur,2K], B = tT [d,2K], both NT) ----
template <int MB, int NB, int K>
__global__ void __launch_bounds__(256, 1)
gemm_out_kernel(const bf16_t* __restrict__ pThl, const bf16_t* __restrict__ tThl,
                float* __restrict__ O, int M, int N)
{
  const int wave = (int)((blockIdx.x * blockDim.x + threadIdx.x) >> 5);
  const int lane = (int)(threadIdx.x & 31);
  const int stripsN = (N >> 4) / NB;
  const int nBlocks = ((M >> 4) / MB) * stripsN;
  if (wave >= nBlocks) return;
  const int tm = wave / stripsN, tn = wave - tm * stripsN;
  const int row0 = tm * MB * 16, col0 = tn * NB * 16;

  v8f acc[MB][NB];
  zero_acc<MB, NB>(acc);
  block_mm_nt<MB, NB, K>(pThl + (size_t)row0 * (2 * K),
                         tThl + (size_t)col0 * (2 * K), lane, acc);

  const int g = lane >> 4, n = lane & 15;
  #pragma unroll
  for (int j = 0; j < MB; ++j) {
    const int row = row0 + j * 16;
    #pragma unroll
    for (int i = 0; i < NB; ++i) {
      const int col = col0 + i * 16;
      #pragma unroll
      for (int v = 0; v < 8; ++v) {
        const int m = v + (g << 3);
        O[(size_t)(row + m) * N + (col + n)] = acc[j][i][v];
      }
    }
  }
}

// --- prep: fp32 -> interleaved [hi(d)|lo(d)] rows (+ optional row norms) ---
__global__ void prep_rows_kernel(const float* __restrict__ X, int d,
                                 bf16_t* __restrict__ out,
                                 float* __restrict__ norms)
{
  __shared__ float red[256];
  const int row = blockIdx.x, t = threadIdx.x;
  const float* x = X + (size_t)row * d;
  bf16_t* o = out + (size_t)row * 2 * d;
  float s = 0.f;
  for (int j = t; j < d; j += 256) {
    const float v = x[j];
    const bf16_t h = (bf16_t)v;
    o[j] = h;
    o[d + j] = (bf16_t)(v - (float)h);
    s += v * v;
  }
  if (norms == nullptr) return;
  red[t] = s; __syncthreads();
  for (int o2 = 128; o2 > 0; o2 >>= 1) { if (t < o2) red[t] += red[t + o2]; __syncthreads(); }
  if (t == 0) norms[row] = red[0];
}

// --- row softmax over S[i,:], write p TRANSPOSED, interleaved hi|lo --------
__global__ void softmaxT_kernel(const float* __restrict__ S, int N, int M,
                                bf16_t* __restrict__ pThl)
{
  __shared__ float red[256];
  const int i = blockIdx.x, t = threadIdx.x;
  const float* row = S + (size_t)i * N;

  float mx = -3.4e38f;
  for (int j = t; j < N; j += 256) mx = fmaxf(mx, row[j]);
  red[t] = mx; __syncthreads();
  for (int o = 128; o > 0; o >>= 1) { if (t < o) red[t] = fmaxf(red[t], red[t + o]); __syncthreads(); }
  mx = red[0]; __syncthreads();

  float sum = 0.f;
  for (int j = t; j < N; j += 256) sum += __expf(row[j] - mx);
  red[t] = sum; __syncthreads();
  for (int o = 128; o > 0; o >>= 1) { if (t < o) red[t] += red[t + o]; __syncthreads(); }
  const float inv = 1.0f / red[0];

  for (int j = t; j < N; j += 256) {
    const float p = __expf(row[j] - mx) * inv;
    const bf16_t h = (bf16_t)p;
    const size_t idx = (size_t)j * (2 * M) + i;    // [N, 2M] rows hi|lo
    pThl[idx] = h;
    pThl[idx + M] = (bf16_t)(p - (float)h);
  }
}

// ---------------------------------------------------------------------------
extern "C" void kernel_launch(void* const* d_in, const int* in_sizes, int n_in,
                              void* d_out, int out_size, void* d_ws, size_t ws_size,
                              hipStream_t stream)
{
  (void)in_sizes; (void)n_in; (void)out_size; (void)ws_size;
  const int d = 1024;
  const int n0 = 16384, n1 = 2048, n2 = 256, n3 = 32;

  const float* h0 = (const float*)d_in[0];
  const float* centers[3] = {(const float*)d_in[1], (const float*)d_in[2], (const float*)d_in[3]};
  const float* Wmat[3]    = {(const float*)d_in[4], (const float*)d_in[6], (const float*)d_in[8]};
  const float* bias[3]    = {(const float*)d_in[5], (const float*)d_in[7], (const float*)d_in[9]};
  const int nprev[3] = {n0, n1, n2};
  const int ncur[3]  = {n1, n2, n3};

  // Workspace carving (256B aligned).
  char* ws = (char*)d_ws;
  size_t off = 0;
  auto alloc = [&](size_t bytes) -> char* {
    off = (off + 255) & ~(size_t)255;
    char* p = ws + off;
    off += bytes;
    return p;
  };

  bf16_t* Hhl = (bf16_t*)alloc((size_t)n0 * 2 * d * 2);   // 64 MB
  bf16_t* Chl = (bf16_t*)alloc((size_t)n1 * 2 * d * 2);   //  8 MB
  bf16_t* Whl = (bf16_t*)alloc((size_t)d * 2 * d * 2);    //  4 MB
  float*  hn  = (float*) alloc((size_t)n0 * 4);
  float*  cn  = (float*) alloc((size_t)n1 * 4);
  float*  S   = (float*) alloc((size_t)n0 * n1 * 4);      // 134 MB logits
  bf16_t* pThl = (bf16_t*)alloc((size_t)n1 * 2 * n0 * 2); // 134 MB
  float*  Hnext = (float*)alloc((size_t)n1 * d * 4);      //   8 MB
  // tT aliased into S: S is dead once softmaxT has consumed it (64 MB <= 134 MB).
  bf16_t* tThl = (bf16_t*)S;

  const float* hsrc = h0;
  for (int l = 0; l < 3; ++l) {
    const int M = nprev[l];   // rows of h at this level
    const int N = ncur[l];    // number of centers

    prep_rows_kernel<<<M, 256, 0, stream>>>(hsrc, d, Hhl, hn);
    prep_rows_kernel<<<N, 256, 0, stream>>>(centers[l], d, Chl, cn);
    prep_rows_kernel<<<d, 256, 0, stream>>>(Wmat[l], d, Whl, nullptr);

    // logits GEMM: N is 2048/256 (NB=4) or 32 (NB=2); M always multiple of 32
    if (((N >> 4) & 3) == 0) {
      const int blocksW = ((M >> 4) / 2) * ((N >> 4) / 4);
      gemm_logits_kernel<2, 4, 1024><<<(blocksW * 32 + 255) / 256, 256, 0, stream>>>(
          Hhl, Chl, hn, cn, S, M, N);
    } else {
      const int blocksW = ((M >> 4) / 2) * ((N >> 4) / 2);
      gemm_logits_kernel<2, 2, 1024><<<(blocksW * 32 + 255) / 256, 256, 0, stream>>>(
          Hhl, Chl, hn, cn, S, M, N);
    }

    softmaxT_kernel<<<M, 256, 0, stream>>>(S, N, M, pThl);

    {
      const int blocksW = ((M >> 4) / 2) * ((d >> 4) / 4);
      gemm_tanh_kernel<2, 4, 1024><<<(blocksW * 32 + 255) / 256, 256, 0, stream>>>(
          Hhl, Whl, bias[l], tThl, M, d);
    }

    float* Out = (l == 2) ? (float*)d_out : Hnext;
    const int blocksW = ((N >> 4) / 2) * ((d >> 4) / 4);
    const int grid = (blocksW * 32 + 255) / 256;
    if (l == 0)
      gemm_out_kernel<2, 4, 16384><<<grid, 256, 0, stream>>>(pThl, tThl, Out, N, d);
    else if (l == 1)
      gemm_out_kernel<2, 4, 2048><<<grid, 256, 0, stream>>>(pThl, tThl, Out, N, d);
    else
      gemm_out_kernel<2, 4, 256><<<grid, 256, 0, stream>>>(pThl, tThl, Out, N, d);

    hsrc = Hnext;   // safe: consumed by prep before being overwritten next level
  }
}
